// CrossViewAttention_27101243638324
// MI455X (gfx1250) — matrix operand order repaired and checked by reading, not compiled, not run on hardware
//
#include <hip/hip_runtime.h>
#include <stdint.h>

// ---------------------------------------------------------------------------
// CrossViewAttention for MI455X (gfx1250, wave32, WMMA + async global->LDS)
//   B=16384, CUR=FEAT=2048, HEADS=16, HEAD_DIM=128
// ---------------------------------------------------------------------------

#define BDIM   16384
#define KDIM   2048
#define NDIM   2048
#define HEADS  16
#define HDIM   128
#define LN_EPS 1e-5f
#define LDSS   40          // LDS row stride in halfs: 80B, 16B-aligned, bank-conflict-free

typedef _Float16 h4   __attribute__((ext_vector_type(4)));
typedef _Float16 h8   __attribute__((ext_vector_type(8)));
typedef _Float16 v16h __attribute__((ext_vector_type(16)));
typedef float    v8f  __attribute__((ext_vector_type(8)));

__device__ __forceinline__ h4 cvt4(float4 f) {
    h4 h;
    h[0] = (_Float16)f.x; h[1] = (_Float16)f.y;
    h[2] = (_Float16)f.z; h[3] = (_Float16)f.w;
    return h;
}

__device__ __forceinline__ v16h make16(h8 lo, h8 hi) {
    union { v16h v; h8 h[2]; } u;
    u.h[0] = lo; u.h[1] = hi;
    return u.v;
}

__device__ __forceinline__ void wait_async0() {
    asm volatile("s_wait_asynccnt 0x0" ::: "memory");
}

// Async DMA of one 128x32-half tile pair (A rows / B weight rows) into LDS.
// 4 x b128 chunks per operand per thread; tracked on ASYNCcnt.
__device__ __forceinline__ void async_copy_tile(const _Float16* __restrict__ X,
                                                const _Float16* __restrict__ Wh,
                                                int m0, int n0, int kk, int tid,
                                                _Float16 (*As)[LDSS],
                                                _Float16 (*Bs)[LDSS])
{
    #pragma unroll
    for (int i = 0; i < 4; ++i) {
        int idx = i * 128 + tid;
        int r = idx >> 2, c = (idx & 3) * 8;
        const _Float16* ga = X + (size_t)(m0 + r) * KDIM + kk + c;
        uint32_t la = (uint32_t)(uintptr_t)&As[r][c];
        asm volatile("global_load_async_to_lds_b128 %0, %1, off"
                     :: "v"(la), "v"(ga) : "memory");
        const _Float16* gb = Wh + (size_t)(n0 + r) * KDIM + kk + c;
        uint32_t lb = (uint32_t)(uintptr_t)&Bs[r][c];
        asm volatile("global_load_async_to_lds_b128 %0, %1, off"
                     :: "v"(lb), "v"(gb) : "memory");
    }
}

// ---------------------------------------------------------------------------
// fp32 -> f16 pre-conversion (one-time pass; GEMM hot loops are pure f16)
// ---------------------------------------------------------------------------
__global__ __launch_bounds__(256)
void cva_cvt_f16(const float* __restrict__ x, _Float16* __restrict__ y, int n4)
{
    int i = blockIdx.x * 256 + threadIdx.x;
    if (i < n4) {
        float4 f = ((const float4*)x)[i];
        *(h4*)(y + (size_t)i * 4) = cvt4(f);
    }
}

// ---------------------------------------------------------------------------
// WMMA GEMM: Y = X(f16)[M,K] * Wh(f16)[N,K]^T (+bias). Block 128 thr (4 waves),
// block tile 128x128, wave tile 64x64 = 16x v_wmma_f32_16x16x32_f16 / K-step.
// Tiles DMA'd global->LDS with async loads, double-buffered.
// FINAL=false: f16 out.  FINAL=true: f32 out + bias + residual (pre-LN y).
// ---------------------------------------------------------------------------
template <bool FINAL>
__global__ __launch_bounds__(128)
void cva_gemm_wmma(const _Float16* __restrict__ X, const _Float16* __restrict__ Wh,
                   const float* __restrict__ bias, const float* __restrict__ resid,
                   float* __restrict__ Yf, _Float16* __restrict__ Yh)
{
    __shared__ _Float16 As[2][128][LDSS];
    __shared__ _Float16 Bs[2][128][LDSS];

    const int tid  = threadIdx.x;
    const int lane = tid & 31;
    const int wid  = tid >> 5;
    const int lr   = lane & 15;     // row (A) / out-col (B) within 16
    const int lh   = lane >> 4;     // half-select
    const int m0   = blockIdx.y * 128;
    const int n0   = blockIdx.x * 128;
    const int wm   = (wid >> 1) * 64;
    const int wn   = (wid & 1) * 64;

    v8f acc[4][4];
    #pragma unroll
    for (int i = 0; i < 4; ++i)
        #pragma unroll
        for (int j = 0; j < 4; ++j)
            #pragma unroll
            for (int e = 0; e < 8; ++e) acc[i][j][e] = 0.0f;

    // prologue: DMA tile 0 into buffer 0
    async_copy_tile(X, Wh, m0, n0, 0, tid, As[0], Bs[0]);
    wait_async0();
    __syncthreads();

    const int KT = KDIM / 32;
    for (int kt = 0; kt < KT; ++kt) {
        const int cur = kt & 1, nxt = cur ^ 1;

        if (kt + 1 < KT)   // DMA next tile while we compute on this one
            async_copy_tile(X, Wh, m0, n0, (kt + 1) * 32, tid, As[nxt], Bs[nxt]);

        // fragments from LDS (ISA wave32 VGPR layouts)
        v16h af[4], bf[4];
        #pragma unroll
        for (int mi = 0; mi < 4; ++mi) {
            const _Float16* p = &As[cur][wm + mi * 16 + lr][0];
            af[mi] = make16(*(const h8*)(p + lh * 8), *(const h8*)(p + 16 + lh * 8));
        }
        #pragma unroll
        for (int ni = 0; ni < 4; ++ni) {
            const _Float16* p = &Bs[cur][wn + ni * 16 + lr][0];
            bf[ni] = make16(*(const h8*)(p + lh * 8), *(const h8*)(p + 16 + lh * 8));
        }

        #pragma unroll
        for (int mi = 0; mi < 4; ++mi)
            #pragma unroll
            for (int ni = 0; ni < 4; ++ni)
                acc[mi][ni] = __builtin_amdgcn_wmma_f32_16x16x32_f16(
                    false, af[mi], false, bf[ni], (short)0, acc[mi][ni], false, false);

        wait_async0();       // next buffer landed in LDS (overlapped with WMMAs)
        __syncthreads();
    }

    // epilogue: C/D layout -> lane(0-15): M=j; lane(16-31): M=8+j; N = lr
    #pragma unroll
    for (int mi = 0; mi < 4; ++mi)
        #pragma unroll
        for (int ni = 0; ni < 4; ++ni) {
            int n = n0 + wn + ni * 16 + lr;
            float bv = bias[n];
            #pragma unroll
            for (int j = 0; j < 8; ++j) {
                int m = m0 + wm + mi * 16 + lh * 8 + j;
                size_t o = (size_t)m * NDIM + n;
                if (FINAL) {
                    Yf[o] = acc[mi][ni][j] + bv + resid[o];
                } else {
                    Yh[o] = (_Float16)(acc[mi][ni][j] + bv);
                }
            }
        }
}

// ---------------------------------------------------------------------------
// Attention: one wave32 per (batch, head); 2-key softmax.
// ---------------------------------------------------------------------------
__global__ __launch_bounds__(256)
void cva_attn(const _Float16* __restrict__ q,  const _Float16* __restrict__ k2,
              const _Float16* __restrict__ k3, const _Float16* __restrict__ v2,
              const _Float16* __restrict__ v3, _Float16* __restrict__ ao,
              float* __restrict__ attnw)
{
    const int wg   = blockIdx.x * 8 + (threadIdx.x >> 5);
    const int lane = threadIdx.x & 31;
    const int b = wg >> 4, h = wg & (HEADS - 1);
    const size_t base = (size_t)b * NDIM + h * HDIM + lane * 4;

    h4 qv  = *(const h4*)(q  + base);
    h4 k2v = *(const h4*)(k2 + base);
    h4 k3v = *(const h4*)(k3 + base);
    h4 v2v = *(const h4*)(v2 + base);
    h4 v3v = *(const h4*)(v3 + base);

    float s2 = 0.f, s3 = 0.f;
    #pragma unroll
    for (int j = 0; j < 4; ++j) {
        float qf = (float)qv[j];
        s2 += qf * (float)k2v[j];
        s3 += qf * (float)k3v[j];
    }
    #pragma unroll
    for (int off = 16; off > 0; off >>= 1) {
        s2 += __shfl_xor(s2, off, 32);
        s3 += __shfl_xor(s3, off, 32);
    }
    const float scale = 0.0883883476483184f;   // 1/sqrt(128)
    float a2 = s2 * scale, a3 = s3 * scale;
    float mx = fmaxf(a2, a3);
    float e2 = __expf(a2 - mx), e3 = __expf(a3 - mx);
    float inv = 1.0f / (e2 + e3);
    float w2 = e2 * inv, w3 = e3 * inv;

    h4 r;
    #pragma unroll
    for (int j = 0; j < 4; ++j)
        r[j] = (_Float16)(w2 * (float)v2v[j] + w3 * (float)v3v[j]);
    *(h4*)(ao + base) = r;

    if (lane == 0) {
        attnw[(size_t)(b * HEADS + h) * 2 + 0] = w2;
        attnw[(size_t)(b * HEADS + h) * 2 + 1] = w3;
    }
}

// ---------------------------------------------------------------------------
// LayerNorm over rows of 2048, in place on d_out.
// ---------------------------------------------------------------------------
__global__ __launch_bounds__(256)
void cva_layernorm(float* __restrict__ y, const float* __restrict__ w,
                   const float* __restrict__ b)
{
    const int row = blockIdx.x, tid = threadIdx.x;
    const int lane = tid & 31, wid = tid >> 5;
    const size_t base = (size_t)row * NDIM;

    float v[8], s = 0.f, ss = 0.f;
    #pragma unroll
    for (int i = 0; i < 8; ++i) {
        int c = tid + i * 256;
        v[i] = y[base + c];
        s += v[i]; ss += v[i] * v[i];
    }
    #pragma unroll
    for (int off = 16; off > 0; off >>= 1) {
        s  += __shfl_xor(s,  off, 32);
        ss += __shfl_xor(ss, off, 32);
    }
    __shared__ float rs[8], rss[8];
    if (lane == 0) { rs[wid] = s; rss[wid] = ss; }
    __syncthreads();
    float ts = 0.f, tss = 0.f;
    #pragma unroll
    for (int j = 0; j < 8; ++j) { ts += rs[j]; tss += rss[j]; }
    const float mean = ts * (1.0f / NDIM);
    const float var  = tss * (1.0f / NDIM) - mean * mean;
    const float rstd = rsqrtf(var + LN_EPS);
    #pragma unroll
    for (int i = 0; i < 8; ++i) {
        int c = tid + i * 256;
        y[base + c] = w[c] * (v[i] - mean) * rstd + b[c];
    }
}

// ---------------------------------------------------------------------------
extern "C" void kernel_launch(void* const* d_in, const int* in_sizes, int n_in,
                              void* d_out, int out_size, void* d_ws, size_t ws_size,
                              hipStream_t stream) {
    (void)in_sizes; (void)n_in; (void)out_size; (void)ws_size;

    const float* cur  = (const float*)d_in[0];
    const float* v2s  = (const float*)d_in[1];
    const float* v3s  = (const float*)d_in[2];
    const float* Wq   = (const float*)d_in[3];
    const float* bq   = (const float*)d_in[4];
    const float* Wk2  = (const float*)d_in[5];
    const float* bk2  = (const float*)d_in[6];
    const float* Wk3  = (const float*)d_in[7];
    const float* bk3  = (const float*)d_in[8];
    const float* Wv2  = (const float*)d_in[9];
    const float* bv2  = (const float*)d_in[10];
    const float* Wv3  = (const float*)d_in[11];
    const float* bv3  = (const float*)d_in[12];
    const float* Wo   = (const float*)d_in[13];
    const float* bo   = (const float*)d_in[14];
    const float* lnw  = (const float*)d_in[15];
    const float* lnb  = (const float*)d_in[16];

    float* out = (float*)d_out;
    const size_t S  = (size_t)BDIM * NDIM;     // 16384*2048 activation elems
    const size_t WS = (size_t)NDIM * KDIM;     // 2048*2048 weight elems
    float* attnw = out + S;                    // [B, H, 1, 2] tail of d_out

    _Float16* wsh = (_Float16*)d_ws;           // (8*S + 6*WS)*2B ~= 560 MiB
    _Float16* qh   = wsh + 0 * S;
    _Float16* k2h  = wsh + 1 * S;
    _Float16* k3h  = wsh + 2 * S;
    _Float16* v2h  = wsh + 3 * S;
    _Float16* v3h  = wsh + 4 * S;
    _Float16* xcur = wsh + 5 * S;              // f16 current_view; reused as attn_out
    _Float16* xv2  = wsh + 6 * S;
    _Float16* xv3  = wsh + 7 * S;
    _Float16* wqh  = wsh + 8 * S + 0 * WS;
    _Float16* wk2h = wsh + 8 * S + 1 * WS;
    _Float16* wk3h = wsh + 8 * S + 2 * WS;
    _Float16* wv2h = wsh + 8 * S + 3 * WS;
    _Float16* wv3h = wsh + 8 * S + 4 * WS;
    _Float16* woh  = wsh + 8 * S + 5 * WS;
    _Float16* aoh  = xcur;                     // alias: xcur dead after q-GEMM

    // 1) one-time fp32 -> f16 conversion (memory-bound, ~30us)
    const int nA4 = (int)(S / 4), nW4 = (int)(WS / 4);
    cva_cvt_f16<<<(nA4 + 255) / 256, 256, 0, stream>>>(cur, xcur, nA4);
    cva_cvt_f16<<<(nA4 + 255) / 256, 256, 0, stream>>>(v2s, xv2,  nA4);
    cva_cvt_f16<<<(nA4 + 255) / 256, 256, 0, stream>>>(v3s, xv3,  nA4);
    cva_cvt_f16<<<(nW4 + 255) / 256, 256, 0, stream>>>(Wq,  wqh,  nW4);
    cva_cvt_f16<<<(nW4 + 255) / 256, 256, 0, stream>>>(Wk2, wk2h, nW4);
    cva_cvt_f16<<<(nW4 + 255) / 256, 256, 0, stream>>>(Wk3, wk3h, nW4);
    cva_cvt_f16<<<(nW4 + 255) / 256, 256, 0, stream>>>(Wv2, wv2h, nW4);
    cva_cvt_f16<<<(nW4 + 255) / 256, 256, 0, stream>>>(Wv3, wv3h, nW4);
    cva_cvt_f16<<<(nW4 + 255) / 256, 256, 0, stream>>>(Wo,  woh,  nW4);

    dim3 gg(NDIM / 128, BDIM / 128);           // (16, 128)
    dim3 gb(128);

    // 2) five projection GEMMs (WMMA, async LDS tiles)
    cva_gemm_wmma<false><<<gg, gb, 0, stream>>>(xcur, wqh,  bq,  nullptr, nullptr, qh);
    cva_gemm_wmma<false><<<gg, gb, 0, stream>>>(xv2,  wk2h, bk2, nullptr, nullptr, k2h);
    cva_gemm_wmma<false><<<gg, gb, 0, stream>>>(xv3,  wk3h, bk3, nullptr, nullptr, k3h);
    cva_gemm_wmma<false><<<gg, gb, 0, stream>>>(xv2,  wv2h, bv2, nullptr, nullptr, v2h);
    cva_gemm_wmma<false><<<gg, gb, 0, stream>>>(xv3,  wv3h, bv3, nullptr, nullptr, v3h);

    // 3) 2-key attention (wave per (b,h)); weights go straight into d_out tail
    cva_attn<<<(BDIM * HEADS) / 8, 256, 0, stream>>>(qh, k2h, k3h, v2h, v3h, aoh, attnw);

    // 4) output GEMM with fused bias + residual -> pre-LN y in d_out
    cva_gemm_wmma<true><<<gg, gb, 0, stream>>>(aoh, woh, bo, cur, out, nullptr);

    // 5) LayerNorm in place
    cva_layernorm<<<BDIM, 256, 0, stream>>>(out, lnw, lnb);
}
